// MultiScaledConvHead_62130996904382
// MI455X (gfx1250) — matrix operand order repaired and checked by reading, not compiled, not run on hardware
//
#include <hip/hip_runtime.h>
#include <math.h>

typedef __attribute__((ext_vector_type(16))) _Float16 v16h;
typedef __attribute__((ext_vector_type(8)))  _Float16 v8h;
typedef __attribute__((ext_vector_type(8)))  float    v8f;

constexpr int B  = 4, S = 1024, D = 1024, H = 16, NM = 4, R = 64, DH = 64;
constexpr int LDM = 72;                 // padded LDS row stride in halfs (144 B)
constexpr float EPS = 1e-8f;

#define DEVI __device__ __forceinline__

// ---- CDNA5 async global->LDS copy (ASYNCcnt-tracked), GVS addressing ----
// LDS byte address comes from a VGPR; global address = SGPR64 base + VGPR32 offset
DEVI unsigned lds_addr_of(const void* p) {
  return (unsigned)(uintptr_t)p;   // flat->LDS uses addr[31:0] (ISA 10.2)
}
DEVI void async_copy16(unsigned lds_byte_off, const void* sbase, unsigned gbyte_off) {
  asm volatile("global_load_async_to_lds_b128 %0, %1, %2"
               :: "v"(lds_byte_off), "v"(gbyte_off), "s"(sbase) : "memory");
}
DEVI void wait_async() {
  asm volatile("s_wait_asynccnt 0" ::: "memory");
}

// ---- WMMA fragment loaders (ISA 7.12.2 wave32 layouts) -----------------
DEVI v16h load_frag_a(const _Float16* base) {
  const int lane = threadIdx.x & 31;
  const _Float16* p = base + (lane & 15) * LDM + ((lane >> 4) << 3);
  v8h lo = *(const v8h*)(p);
  v8h hi = *(const v8h*)(p + 16);
  return __builtin_shufflevector(lo, hi, 0,1,2,3,4,5,6,7,8,9,10,11,12,13,14,15);
}
DEVI v16h load_frag_b(const _Float16* base) {
  const int lane = threadIdx.x & 31;
  const _Float16* p = base + (lane & 15) * LDM + ((lane >> 4) << 3);
  v8h lo = *(const v8h*)(p);
  v8h hi = *(const v8h*)(p + 16);
  return __builtin_shufflevector(lo, hi, 0,1,2,3,4,5,6,7,8,9,10,11,12,13,14,15);
}
DEVI v8f wmma_f32(v16h a, v16h b, v8f c) {
  return __builtin_amdgcn_wmma_f32_16x16x32_f16(false, a, false, b, (short)0, c,
                                                false, false);
}

// ---- f32 -> f16 conversion --------------------------------------------
__global__ void cvt_f16_kernel(const float* __restrict__ in,
                               _Float16* __restrict__ out, int n) {
  int i = blockIdx.x * 256 + threadIdx.x;
  if (i < n) out[i] = (_Float16)in[i];
}

// ---- projections: P = x @ W^T + b, stored f16, natural [B,S,D] layout --
__global__ __launch_bounds__(128) void proj_gemm_kernel(
    const _Float16* __restrict__ xh,
    const _Float16* __restrict__ wq, const _Float16* __restrict__ wk,
    const _Float16* __restrict__ wv,
    const float* __restrict__ bq, const float* __restrict__ bk,
    const float* __restrict__ bv,
    _Float16* __restrict__ qh, _Float16* __restrict__ kh,
    _Float16* __restrict__ vh) {
  __shared__ _Float16 As[64 * LDM];
  __shared__ _Float16 Bs[64 * LDM];
  const int tid = threadIdx.x, lane = tid & 31, wave = tid >> 5;
  const int hi = lane >> 4, ln = lane & 15;
  const int j0 = blockIdx.x * 64;          // output-feature tile
  const int n0 = blockIdx.y * 64;          // row tile over B*S
  const int z  = blockIdx.z;               // 0=q 1=k 2=v
  const _Float16* W    = (z == 0) ? wq : (z == 1) ? wk : wv;
  const float*    bias = (z == 0) ? bq : (z == 1) ? bk : bv;
  _Float16*       outp = (z == 0) ? qh : (z == 1) ? kh : vh;
  const int m0 = wave * 16;

  v8f c0 = {}, c1 = {}, c2 = {}, c3 = {};
  for (int kk = 0; kk < D; kk += 64) {
    __syncthreads();
    // async global->LDS staging: 512 chunks of 16 B per tile
    for (int i = tid; i < 512; i += 128) {
      int r = i >> 3, c = (i & 7) * 8;
      async_copy16(lds_addr_of(&As[r * LDM + c]), xh,
                   (unsigned)(((n0 + r) * D + kk + c) * 2));
      async_copy16(lds_addr_of(&Bs[r * LDM + c]), W,
                   (unsigned)(((j0 + r) * D + kk + c) * 2));
    }
    wait_async();
    __syncthreads();

    // preload all fragments, then back-to-back WMMAs
    v16h a0 = load_frag_a(&As[m0 * LDM + 0]);
    v16h a1 = load_frag_a(&As[m0 * LDM + 32]);
    v16h bfr[8];
#pragma unroll
    for (int n = 0; n < 4; n++) {
      bfr[2 * n]     = load_frag_b(&Bs[(n * 16) * LDM + 0]);
      bfr[2 * n + 1] = load_frag_b(&Bs[(n * 16) * LDM + 32]);
    }
    c0 = wmma_f32(a0, bfr[0], c0); c0 = wmma_f32(a1, bfr[1], c0);
    c1 = wmma_f32(a0, bfr[2], c1); c1 = wmma_f32(a1, bfr[3], c1);
    c2 = wmma_f32(a0, bfr[4], c2); c2 = wmma_f32(a1, bfr[5], c2);
    c3 = wmma_f32(a0, bfr[6], c3); c3 = wmma_f32(a1, bfr[7], c3);
  }
#pragma unroll
  for (int r = 0; r < 8; r++) {
    size_t ro = (size_t)(n0 + m0 + r + hi * 8) * D + j0;
    outp[ro +  0 + ln] = (_Float16)(c0[r] + bias[j0 +  0 + ln]);
    outp[ro + 16 + ln] = (_Float16)(c1[r] + bias[j0 + 16 + ln]);
    outp[ro + 32 + ln] = (_Float16)(c2[r] + bias[j0 + 32 + ln]);
    outp[ro + 48 + ln] = (_Float16)(c3[r] + bias[j0 + 48 + ln]);
  }
}

// ---- low-rank mask LM[m,s,t] = sum_r m1[m,r,s]*m2[m,r,t] (fp32) --------
__global__ void lowrank_mask_kernel(const float* __restrict__ m1,
                                    const float* __restrict__ m2,
                                    float* __restrict__ LM) {
  const int m = blockIdx.z, s = blockIdx.y;
  const int t = blockIdx.x * 256 + threadIdx.x;
  float acc = 0.f;
#pragma unroll 8
  for (int r = 0; r < R; r++)
    acc += m1[((size_t)m * R + r) * S + s] * m2[((size_t)m * R + r) * S + t];
  LM[((size_t)m * S + s) * S + t] = acc;
}

// ---- fused attention: O_unnorm = (QK^T/8 * gather(LM)) @ V ; fro partials
__global__ __launch_bounds__(128) void attn_kernel(
    const _Float16* __restrict__ qh, const _Float16* __restrict__ kh,
    const _Float16* __restrict__ vh, const float* __restrict__ LM,
    const long long* __restrict__ rbk, float* __restrict__ O,
    float* __restrict__ froP) {
  __shared__ _Float16 Qs[64 * LDM];
  __shared__ _Float16 Ks[64 * LDM];   // [t][dh]
  __shared__ _Float16 Vt[64 * LDM];   // transposed: [dh][t]
  __shared__ _Float16 Asm[64 * LDM];  // masked att tile, f16
  __shared__ float red[128];

  const int tid = threadIdx.x, lane = tid & 31, wave = tid >> 5;
  const int hi = lane >> 4, ln = lane & 15;
  const int s0 = blockIdx.x * 64;
  const int bh = blockIdx.y;
  const int b = bh / H, h = bh % H, mi = h % NM;
  const int sm0 = wave * 16;

  const _Float16* qbase = qh + (size_t)b * S * D + h * DH;
  const _Float16* kbase = kh + (size_t)b * S * D + h * DH;
  const _Float16* vbase = vh + (size_t)b * S * D + h * DH;

  // stage Q tile once (async), then keep its fragments in registers forever
  for (int i = tid; i < 512; i += 128) {
    int r = i >> 3, c = (i & 7) * 8;
    async_copy16(lds_addr_of(&Qs[r * LDM + c]), qbase,
                 (unsigned)(((s0 + r) * D + c) * 2));
  }
  wait_async();
  __syncthreads();
  const v16h qa0 = load_frag_a(&Qs[sm0 * LDM + 0]);
  const v16h qa1 = load_frag_a(&Qs[sm0 * LDM + 32]);

  v8f o0 = {}, o1 = {}, o2 = {}, o3 = {};
  float fro = 0.f;

  for (int t0 = 0; t0 < S; t0 += 64) {
    __syncthreads();   // previous iteration done reading Ks/Vt/Asm
    // K tile: async direct-to-LDS ; V tile: register transpose into LDS
    for (int i = tid; i < 512; i += 128) {
      int r = i >> 3, c = (i & 7) * 8;
      async_copy16(lds_addr_of(&Ks[r * LDM + c]), kbase,
                   (unsigned)(((t0 + r) * D + c) * 2));
      v8h vv = *(const v8h*)(vbase + (size_t)(t0 + r) * D + c);
#pragma unroll
      for (int j = 0; j < 8; j++) Vt[(c + j) * LDM + r] = vv[j];
    }
    if (t0 + 64 < S)   // global_prefetch_b8 for next K tile
      __builtin_prefetch(kbase + (size_t)(t0 + 64 + (tid >> 1)) * D + (tid & 1) * 32, 0, 1);
    wait_async();
    __syncthreads();

    // S-tile: preload the 8 K fragments, then 8 WMMAs + gather/mask/fro
    v16h kb[8];
#pragma unroll
    for (int tsub = 0; tsub < 4; tsub++) {
      kb[2 * tsub]     = load_frag_b(&Ks[(tsub * 16) * LDM + 0]);
      kb[2 * tsub + 1] = load_frag_b(&Ks[(tsub * 16) * LDM + 32]);
    }
#pragma unroll
    for (int tsub = 0; tsub < 4; tsub++) {
      v8f acc = {};
      acc = wmma_f32(qa0, kb[2 * tsub], acc);
      acc = wmma_f32(qa1, kb[2 * tsub + 1], acc);
      const int tglob = t0 + tsub * 16 + ln;
#pragma unroll
      for (int r = 0; r < 8; r++) {
        const int sloc = sm0 + r + hi * 8;
        const size_t rowbase = ((size_t)mi * S + (s0 + sloc)) * S;
        const long long g = rbk[rowbase + tglob];
        const float mv = LM[rowbase + (size_t)g];
        const float av = acc[r] * 0.125f * mv;   // 1/sqrt(64)
        fro += av * av;
        Asm[sloc * LDM + tsub * 16 + ln] = (_Float16)av;
      }
    }
    __syncthreads();

    // O += att_tile @ V_tile (K-dim = t, N = dh): preload then 8 WMMAs
    v16h aa0 = load_frag_a(&Asm[sm0 * LDM + 0]);
    v16h aa1 = load_frag_a(&Asm[sm0 * LDM + 32]);
    v16h vb[8];
#pragma unroll
    for (int n = 0; n < 4; n++) {
      vb[2 * n]     = load_frag_b(&Vt[(n * 16) * LDM + 0]);
      vb[2 * n + 1] = load_frag_b(&Vt[(n * 16) * LDM + 32]);
    }
    o0 = wmma_f32(aa0, vb[0], o0); o0 = wmma_f32(aa1, vb[1], o0);
    o1 = wmma_f32(aa0, vb[2], o1); o1 = wmma_f32(aa1, vb[3], o1);
    o2 = wmma_f32(aa0, vb[4], o2); o2 = wmma_f32(aa1, vb[5], o2);
    o3 = wmma_f32(aa0, vb[6], o3); o3 = wmma_f32(aa1, vb[7], o3);
  }

#pragma unroll
  for (int r = 0; r < 8; r++) {
    const size_t ro = ((size_t)bh * S + (s0 + sm0 + r + hi * 8)) * DH;
    O[ro +  0 + ln] = o0[r];
    O[ro + 16 + ln] = o1[r];
    O[ro + 32 + ln] = o2[r];
    O[ro + 48 + ln] = o3[r];
  }

  red[tid] = fro;
  __syncthreads();
  for (int off = 64; off > 0; off >>= 1) {
    if (tid < off) red[tid] += red[tid + off];
    __syncthreads();
  }
  if (tid == 0) froP[bh * 16 + blockIdx.x] = red[0];
}

// ---- finalize: per-head Frobenius scale + [B,H,S,DH] -> [B,S,D] --------
__global__ void finalize_kernel(const float* __restrict__ O,
                                const float* __restrict__ froP,
                                float* __restrict__ out) {
  __shared__ float s_inv;
  const int bh = blockIdx.x;
  if (threadIdx.x == 0) {
    float s = 0.f;
    for (int i = 0; i < 16; i++) s += froP[bh * 16 + i];
    s_inv = 1.0f / (sqrtf(s) + EPS);
  }
  __syncthreads();
  const int b = bh / H, h = bh % H;
  const float inv = s_inv;
  for (int i = threadIdx.x; i < S * DH; i += 256) {
    const int s = i >> 6, dh = i & 63;
    out[((size_t)b * S + s) * D + h * DH + dh] =
        O[((size_t)bh * S + s) * DH + dh] * inv;
  }
}

extern "C" void kernel_launch(void* const* d_in, const int* in_sizes, int n_in,
                              void* d_out, int out_size, void* d_ws, size_t ws_size,
                              hipStream_t stream) {
  const float* x  = (const float*)d_in[0];
  const float* Wk = (const float*)d_in[1];
  const float* bk = (const float*)d_in[2];
  const float* Wq = (const float*)d_in[3];
  const float* bq = (const float*)d_in[4];
  const float* Wv = (const float*)d_in[5];
  const float* bv = (const float*)d_in[6];
  const float* m1 = (const float*)d_in[7];
  const float* m2 = (const float*)d_in[8];
  const long long* rbk = (const long long*)d_in[9];   // int64 roll_back
  float* out = (float*)d_out;
  (void)in_sizes; (void)n_in; (void)out_size; (void)ws_size;

  char* w = (char*)d_ws;
  size_t off = 0;
  auto take = [&](size_t bytes) {
    char* p = w + off;
    off = (off + bytes + 255) & ~(size_t)255;
    return p;
  };
  _Float16* xh  = (_Float16*)take((size_t)B * S * D * 2);   // 8 MB
  _Float16* wqh = (_Float16*)take((size_t)D * D * 2);       // 2 MB
  _Float16* wkh = (_Float16*)take((size_t)D * D * 2);
  _Float16* wvh = (_Float16*)take((size_t)D * D * 2);
  _Float16* qh  = (_Float16*)take((size_t)B * S * D * 2);   // 8 MB each
  _Float16* kh  = (_Float16*)take((size_t)B * S * D * 2);
  _Float16* vh  = (_Float16*)take((size_t)B * S * D * 2);
  float*    LM  = (float*)take((size_t)NM * S * S * 4);     // 16 MB
  float*    O   = (float*)take((size_t)B * H * S * DH * 4); // 16 MB
  float*    frp = (float*)take((size_t)B * H * 16 * 4);

  cvt_f16_kernel<<<(B * S * D + 255) / 256, 256, 0, stream>>>(x, xh, B * S * D);
  cvt_f16_kernel<<<(D * D + 255) / 256, 256, 0, stream>>>(Wq, wqh, D * D);
  cvt_f16_kernel<<<(D * D + 255) / 256, 256, 0, stream>>>(Wk, wkh, D * D);
  cvt_f16_kernel<<<(D * D + 255) / 256, 256, 0, stream>>>(Wv, wvh, D * D);

  proj_gemm_kernel<<<dim3(D / 64, (B * S) / 64, 3), 128, 0, stream>>>(
      xh, wqh, wkh, wvh, bq, bk, bv, qh, kh, vh);

  lowrank_mask_kernel<<<dim3(S / 256, S, NM), 256, 0, stream>>>(m1, m2, LM);

  attn_kernel<<<dim3(S / 64, B * H), 128, 0, stream>>>(qh, kh, vh, LM, rbk, O, frp);

  finalize_kernel<<<dim3(B * H), 256, 0, stream>>>(O, frp, out);
}
